// FrameFusion_17197049053683
// MI455X (gfx1250) — matrix operand order repaired and checked by reading, not compiled, not run on hardware
//
#include <hip/hip_runtime.h>

// Problem constants (fixed by setup_inputs in the reference)
#define B_SZ      2
#define H_SZ      8
#define S_TOK     2176
#define D_DIM     4096
#define IMG_START 64
#define N_IMG     2048
#define TOPK      1024
#define KEEP      1152      // 64 + 1024 + 64
#define NBLK      8         // workgroups cooperating on the sequential DPP loop
#define SLICE     (N_IMG / NBLK)   // 256 columns per DPP block

typedef __attribute__((ext_vector_type(16))) __bf16 v16bf;
typedef __attribute__((ext_vector_type(8)))  float  v8f;

// ---------- helpers ----------

// fp32 -> bf16 round-to-nearest-even
__device__ __forceinline__ unsigned short f2bf(float f) {
  unsigned int u = __float_as_uint(f);
  u += 0x7FFFu + ((u >> 16) & 1u);
  return (unsigned short)(u >> 16);
}

// A fragment, 16x32 bf16 (M x K), from row-major X[row.., k..]
// lane L: m = L&15, h = L>>4; elems 0..7 = K (k + h*8 ..), elems 8..15 = K (k + 16 + h*8 ..)
__device__ __forceinline__ v16bf load_afrag(const unsigned short* __restrict__ X,
                                            int row, int k, int lane) {
  int m = lane & 15, h = lane >> 4;
  const unsigned short* p = X + (size_t)(row + m) * D_DIM + (k + h * 8);
  union { uint4 u[2]; v16bf v; } un;
  un.u[0] = *(const uint4*)p;
  un.u[1] = *(const uint4*)(p + 16);
  return un.v;
}

// B fragment, 32x16 bf16 (K x N), where B[k][n] = X[col+n][k] (Gram: B = X^T tile)
// lane L: n = L&15, h = L>>4; elems 0..7 = K (k + h*8 ..), elems 8..15 = K (k + 16 + h*8 ..)
__device__ __forceinline__ v16bf load_bfrag(const unsigned short* __restrict__ X,
                                            int col, int k, int lane) {
  int n = lane & 15, h = lane >> 4;
  const unsigned short* p = X + (size_t)(col + n) * D_DIM + (k + h * 8);
  union { uint4 u[2]; v16bf v; } un;
  un.u[0] = *(const uint4*)p;
  un.u[1] = *(const uint4*)(p + 16);
  return un.v;
}

// C/D 16x16 f32: lane L holds column n = L&15; element e -> row = 8*(L>>4) + e
__device__ __forceinline__ void store_tile(v8f c, float* __restrict__ K,
                                           const float* __restrict__ rel,
                                           int row, int col, int n, int hh) {
  float rc = rel[col + n];
#pragma unroll
  for (int e = 0; e < 8; ++e) {
    int r = row + hh * 8 + e;
    K[(size_t)r * N_IMG + (col + n)] = c[e] * rel[r] * rc;
  }
}

// resident-grid barrier: release stores, bump counter, acquire-spin to target
__device__ __forceinline__ void grid_barrier(int* cnt, int target) {
  __threadfence();          // each thread's global stores complete
  __syncthreads();
  if (threadIdx.x == 0) {
    __hip_atomic_fetch_add(cnt, 1, __ATOMIC_RELEASE, __HIP_MEMORY_SCOPE_AGENT);
    while (__hip_atomic_load(cnt, __ATOMIC_ACQUIRE, __HIP_MEMORY_SCOPE_AGENT) < target) {
      __builtin_amdgcn_s_sleep(1);
    }
  }
  __syncthreads();          // acquire (WGP$ inv) by wave 0 covers the whole WGP-resident block
}

// ---------- kernels ----------

// zero accumulators / flags / barrier counter (runs every launch -> deterministic)
__global__ __launch_bounds__(256) void k_init(float* attn_sum, int* flags, int* bar_cnt) {
  int t = blockIdx.x * 256 + threadIdx.x;
  if (t < B_SZ * N_IMG) attn_sum[t] = 0.f;
  if (t < N_IMG)        flags[t] = 0;
  if (t == 0)           bar_cnt[0] = 0;
}

// column sums of attn over (H, q) restricted to image columns; 303 MB streaming read
// grid: (8 col-tiles, 16 q-chunks, B*H), block 256
__global__ __launch_bounds__(256) void k_attn(const float* __restrict__ attn,
                                              float* __restrict__ attn_sum) {
  int ct = blockIdx.x, qc = blockIdx.y, bh = blockIdx.z;
  int b = bh / H_SZ;
  int colrel = ct * 256 + threadIdx.x;            // 0..2047
  const float* base = attn + ((size_t)bh * S_TOK + (size_t)qc * 136) * S_TOK
                           + (IMG_START + colrel);
  float s = 0.f;
  for (int q = 0; q < 136; ++q) s += base[(size_t)q * S_TOK];
  atomicAdd(&attn_sum[b * N_IMG + colrel], s);
}

// rel = (-avg - min + 1e-6) / (max - min), min/max global over both batches
__global__ __launch_bounds__(1024) void k_rel(const float* __restrict__ attn_sum,
                                              float* __restrict__ rel) {
  __shared__ float smn[1024], smx[1024];
  const float inv = 1.0f / (float)(H_SZ * S_TOK);
  int tid = threadIdx.x;
  float vals[4];
  float mn = 3.4e38f, mx = -3.4e38f;
#pragma unroll
  for (int i = 0; i < 4; ++i) {
    float v = -attn_sum[i * 1024 + tid] * inv;
    vals[i] = v; mn = fminf(mn, v); mx = fmaxf(mx, v);
  }
  smn[tid] = mn; smx[tid] = mx; __syncthreads();
  for (int s = 512; s > 0; s >>= 1) {
    if (tid < s) { smn[tid] = fminf(smn[tid], smn[tid + s]);
                   smx[tid] = fmaxf(smx[tid], smx[tid + s]); }
    __syncthreads();
  }
  mn = smn[0]; mx = smx[0];
  float sc = 1.0f / (mx - mn);
#pragma unroll
  for (int i = 0; i < 4; ++i) rel[i * 1024 + tid] = (vals[i] - mn + 1e-6f) * sc;
}

// L2-normalize batch-0 image rows (fp32) and store bf16 for WMMA
__global__ __launch_bounds__(256) void k_norm(const float* __restrict__ hs,
                                              unsigned short* __restrict__ xbf) {
  __shared__ float red[256];
  int n = blockIdx.x;                              // 0..2047 (batch 0 only; only sel[:,0] is used)
  int tid = threadIdx.x;
  const float4* row = (const float4*)(hs + (size_t)(IMG_START + n) * D_DIM);
  float4 v[4];
  float ss = 0.f;
#pragma unroll
  for (int i = 0; i < 4; ++i) {
    v[i] = row[i * 256 + tid];
    ss += v[i].x * v[i].x + v[i].y * v[i].y + v[i].z * v[i].z + v[i].w * v[i].w;
  }
  red[tid] = ss; __syncthreads();
  for (int s = 128; s > 0; s >>= 1) { if (tid < s) red[tid] += red[tid + s]; __syncthreads(); }
  float rn = rsqrtf(red[0]);
  uint2* dst = (uint2*)(xbf + (size_t)n * D_DIM);
#pragma unroll
  for (int i = 0; i < 4; ++i) {
    float a0 = v[i].x * rn, a1 = v[i].y * rn, a2 = v[i].z * rn, a3 = v[i].w * rn;
    uint2 pk;
    pk.x = (unsigned)f2bf(a0) | ((unsigned)f2bf(a1) << 16);
    pk.y = (unsigned)f2bf(a2) | ((unsigned)f2bf(a3) << 16);
    dst[i * 256 + tid] = pk;
  }
}

// DPP kernel matrix K = rel_i * (x x^T) * rel_j via bf16 WMMA, f32 accumulate.
// block = 8 waves; wave -> 32x32 output (4 accumulators); block tiles 128 rows x 64 cols.
__global__ __launch_bounds__(256) void k_gram(const unsigned short* __restrict__ X,
                                              const float* __restrict__ rel,
                                              float* __restrict__ K) {
  int lane = threadIdx.x & 31;
  int wave = threadIdx.x >> 5;
  int row0 = blockIdx.y * 128 + (wave & 3) * 32;
  int col0 = blockIdx.x * 64  + (wave >> 2) * 32;
  v8f c00 = {}, c01 = {}, c10 = {}, c11 = {};
  for (int k = 0; k < D_DIM; k += 32) {
    v16bf a0 = load_afrag(X, row0,      k, lane);
    v16bf a1 = load_afrag(X, row0 + 16, k, lane);
    v16bf b0 = load_bfrag(X, col0,      k, lane);
    v16bf b1 = load_bfrag(X, col0 + 16, k, lane);
    c00 = __builtin_amdgcn_wmma_f32_16x16x32_bf16(false, a0, false, b0, (short)0, c00, false, false);
    c01 = __builtin_amdgcn_wmma_f32_16x16x32_bf16(false, a0, false, b1, (short)0, c01, false, false);
    c10 = __builtin_amdgcn_wmma_f32_16x16x32_bf16(false, a1, false, b0, (short)0, c10, false, false);
    c11 = __builtin_amdgcn_wmma_f32_16x16x32_bf16(false, a1, false, b1, (short)0, c11, false, false);
  }
  int n = lane & 15, hh = lane >> 4;
  store_tile(c00, K, rel, row0,      col0,      n, hh);
  store_tile(c01, K, rel, row0,      col0 + 16, n, hh);
  store_tile(c10, K, rel, row0 + 16, col0,      n, hh);
  store_tile(c11, K, rel, row0 + 16, col0 + 16, n, hh);
}

// Greedy DPP: 1024 sequential steps, 8 cooperating blocks each own a 256-column slice.
__global__ __launch_bounds__(SLICE) void k_dpp(const float* __restrict__ K,
                                               float* __restrict__ cis,
                                               int* __restrict__ flags,
                                               float* __restrict__ cand_val,
                                               int* __restrict__ cand_idx,
                                               int* __restrict__ bar_cnt) {
  __shared__ float di2s[SLICE];
  __shared__ float cisj[TOPK];
  __shared__ float rv[SLICE];
  __shared__ int   ri[SLICE];
  int tid = threadIdx.x;
  int blk = blockIdx.x;
  int n = blk * SLICE + tid;                       // this thread's column
  di2s[tid] = K[(size_t)n * N_IMG + n];            // diag(kernel)
  int target = 0;
  for (int i = 0; i < TOPK; ++i) {
    __syncthreads();
    // local argmax over slice (ties -> smallest index, matching jnp.argmax)
    rv[tid] = di2s[tid]; ri[tid] = n;
    __syncthreads();
    for (int s = SLICE / 2; s > 0; s >>= 1) {
      if (tid < s) {
        float v2 = rv[tid + s]; int i2 = ri[tid + s];
        if (v2 > rv[tid] || (v2 == rv[tid] && i2 < ri[tid])) { rv[tid] = v2; ri[tid] = i2; }
      }
      __syncthreads();
    }
    if (tid == 0) { cand_val[blk] = rv[0]; cand_idx[blk] = ri[0]; }
    target += NBLK; grid_barrier(bar_cnt, target);
    // every block combines the 8 candidates identically
    float bv = cand_val[0]; int j = cand_idx[0];
#pragma unroll
    for (int t = 1; t < NBLK; ++t) {
      float v2 = cand_val[t]; int i2 = cand_idx[t];
      if (v2 > bv || (v2 == bv && i2 < j)) { bv = v2; j = i2; }
    }
    float srs = rsqrtf(bv);
    // gather cis[:, j] (written by the block owning column j; visible after barrier)
    for (int t = tid; t < i; t += SLICE) cisj[t] = cis[(size_t)t * N_IMG + j];
    __syncthreads();
    // proj_n = sum_t cisj[t] * cis[t][n]   (rows t<i, self-written slice, coalesced)
    float p = 0.f;
    for (int t = 0; t < i; ++t) p += cisj[t] * cis[(size_t)t * N_IMG + n];
    float e = (K[(size_t)j * N_IMG + n] - p) * srs;
    cis[(size_t)i * N_IMG + n] = e;
    float nd = di2s[tid] - e * e;
    if (n == j) { nd = -__builtin_inff(); flags[j] = 1; }
    di2s[tid] = nd;
    target += NBLK; grid_barrier(bar_cnt, target);   // cis row i visible to all blocks
  }
}

// sorted keep list: [0..63] ++ sorted(selected + 64) ++ [2112..2175]
__global__ __launch_bounds__(1024) void k_compact(const int* __restrict__ flags,
                                                  int* __restrict__ keep) {
  __shared__ int sc[1024];
  int tid = threadIdx.x;
  int f0 = flags[2 * tid], f1 = flags[2 * tid + 1];
  int s = f0 + f1;
  sc[tid] = s; __syncthreads();
  for (int off = 1; off < 1024; off <<= 1) {
    int v = (tid >= off) ? sc[tid - off] : 0;
    __syncthreads();
    sc[tid] += v;
    __syncthreads();
  }
  int excl = sc[tid] - s;                           // selected before column 2*tid
  if (f0) keep[64 + excl]      = IMG_START + 2 * tid;
  if (f1) keep[64 + excl + f0] = IMG_START + 2 * tid + 1;
  if (tid < IMG_START) {
    keep[tid] = tid;
    keep[64 + TOPK + tid] = IMG_START + N_IMG + tid;
  }
}

// gather hs/pos_cos/pos_sin rows for both batches using the shared keep list
__global__ __launch_bounds__(256) void k_gather(const float* __restrict__ hs,
                                                const float* __restrict__ pc,
                                                const float* __restrict__ ps,
                                                const int* __restrict__ keep,
                                                float* __restrict__ out) {
  int r = blockIdx.x % KEEP;
  int b = blockIdx.x / KEEP;
  int src = keep[r];
  const float4* h4 = (const float4*)(hs + ((size_t)b * S_TOK + src) * D_DIM);
  float4* o4 = (float4*)(out + ((size_t)b * KEEP + r) * D_DIM);
  for (int i = threadIdx.x; i < D_DIM / 4; i += 256) o4[i] = h4[i];
  float* pco = out + (size_t)B_SZ * KEEP * D_DIM;
  float* pso = pco + (size_t)B_SZ * KEEP * 128;
  if (threadIdx.x < 128) {
    pco[((size_t)b * KEEP + r) * 128 + threadIdx.x] = pc[((size_t)b * S_TOK + src) * 128 + threadIdx.x];
    pso[((size_t)b * KEEP + r) * 128 + threadIdx.x] = ps[((size_t)b * S_TOK + src) * 128 + threadIdx.x];
  }
}

// ---------- launch ----------

extern "C" void kernel_launch(void* const* d_in, const int* in_sizes, int n_in,
                              void* d_out, int out_size, void* d_ws, size_t ws_size,
                              hipStream_t stream) {
  (void)in_sizes; (void)n_in; (void)out_size; (void)ws_size;
  const float* hs   = (const float*)d_in[0];
  const float* attn = (const float*)d_in[1];
  const float* pc   = (const float*)d_in[2];
  const float* ps   = (const float*)d_in[3];
  float* out = (float*)d_out;

  char* w = (char*)d_ws;
  auto alloc = [&](size_t bytes) -> char* {
    char* p = w; w += (bytes + 255) & ~(size_t)255; return p;
  };
  float*          attn_sum = (float*)alloc((size_t)B_SZ * N_IMG * 4);
  float*          rel      = (float*)alloc((size_t)B_SZ * N_IMG * 4);
  unsigned short* xbf      = (unsigned short*)alloc((size_t)N_IMG * D_DIM * 2);  // 16 MB
  float*          Kmat     = (float*)alloc((size_t)N_IMG * N_IMG * 4);           // 16 MB
  float*          cis      = (float*)alloc((size_t)TOPK * N_IMG * 4);            //  8 MB
  int*            flags    = (int*)alloc((size_t)N_IMG * 4);
  int*            keep     = (int*)alloc((size_t)KEEP * 4);
  float*          cand_val = (float*)alloc((size_t)NBLK * 4);
  int*            cand_idx = (int*)alloc((size_t)NBLK * 4);
  int*            bar_cnt  = (int*)alloc(4);

  k_init   <<<dim3(16), dim3(256), 0, stream>>>(attn_sum, flags, bar_cnt);
  k_attn   <<<dim3(8, 16, B_SZ * H_SZ), dim3(256), 0, stream>>>(attn, attn_sum);
  k_rel    <<<1, 1024, 0, stream>>>(attn_sum, rel);
  k_norm   <<<N_IMG, 256, 0, stream>>>(hs, xbf);
  k_gram   <<<dim3(N_IMG / 64, N_IMG / 128), 256, 0, stream>>>(xbf, rel, Kmat);
  k_dpp    <<<NBLK, SLICE, 0, stream>>>(Kmat, cis, flags, cand_val, cand_idx, bar_cnt);
  k_compact<<<1, 1024, 0, stream>>>(flags, keep);
  k_gather <<<B_SZ * KEEP, 256, 0, stream>>>(hs, pc, ps, keep, out);
}
